// CrossSelfAtt_5549097746878
// MI455X (gfx1250) — compile-verified, matrix operand order checked by reading, not run on hardware
//
#include <hip/hip_runtime.h>
#include <math.h>

// ---------------------------------------------------------------------------
// CDNA5 (gfx1250) CrossSelfAtt: bf16 WMMA GEMMs (fp32 accumulate), operands
// pre-converted to bf16, weights pre-transposed to (N,K). LDS staging via the
// Tensor Data Mover with double-buffered software pipelining
// (tensor_load_to_lds for tile k+1 overlaps WMMA on tile k; s_wait_tensorcnt 2
// exploits in-order TDM completion). Wave32 throughout.
// ---------------------------------------------------------------------------

typedef __attribute__((ext_vector_type(16))) __bf16 v16bf;
typedef __attribute__((ext_vector_type(8)))  __bf16 v8bf;
typedef __attribute__((ext_vector_type(8)))  float  v8f;
typedef __attribute__((ext_vector_type(4)))  unsigned int u32x4;
typedef __attribute__((ext_vector_type(8)))  int i32x8;
typedef __attribute__((ext_vector_type(4)))  int i32x4;

#if __has_builtin(__builtin_amdgcn_tensor_load_to_lds) && \
    __has_builtin(__builtin_amdgcn_s_wait_tensorcnt)
#define USE_TDM 1
#endif

__device__ __forceinline__ __bf16 f2bf(float f) {
    union { float f; unsigned u; } in; in.f = f;
    unsigned u = in.u;
    unsigned r = u + 0x7FFFu + ((u >> 16) & 1u);   // round-to-nearest-even
    union { unsigned short s; __bf16 b; } out;
    out.s = (unsigned short)(r >> 16);
    return out.b;
}

#define LDS_STRIDE 40   // 32 data bf16 + 8 pad = 20 DWORDs per row (16B-aligned)

#ifdef USE_TDM
// TDM 2D tile load: `rows` x 32 bf16 tile from row-major (tensor_rows x K)
// bf16 tensor into LDS with an 8-bf16 (4-DWORD) pad after each 16-DWORD row,
// matching LDS_STRIDE. D# per CDNA5 ISA 8.3-8.5.
__device__ __forceinline__ void tdm_load_tile(const __bf16* gsrc, __bf16* lds,
                                              int rows, int K, int tensor_rows)
{
    unsigned long long ga = (unsigned long long)(uintptr_t)gsrc;
    unsigned lad = (unsigned)(uintptr_t)lds;
    u32x4 g0;
    g0[0] = 1u;                                   // count=1 (valid user D#)
    g0[1] = lad;                                  // lds_addr (bytes)
    g0[2] = (unsigned)ga;                         // global_addr[31:0]
    g0[3] = (unsigned)((ga >> 32) & 0x01FFFFFFu)  // global_addr[56:32]
          | (2u << 30);                           // type=2 ("image")
    unsigned td0 = (unsigned)K;
    unsigned td1 = (unsigned)tensor_rows;
    i32x8 g1;
    g1[0] = (int)((1u << 16)       // data_size = 1 -> 2 bytes
                | (1u << 20)       // pad_enable
                | (3u << 22)       // pad_interval: 16 DWORDs (one 64B row)
                | (3u << 25));     // pad_amount: 4 DWORDs (8 bf16)
    g1[1] = (int)((td0 & 0xFFFFu) << 16);                         // tensor_dim0 lo
    g1[2] = (int)(((td0 >> 16) & 0xFFFFu) | ((td1 & 0xFFFFu) << 16));
    g1[3] = (int)(((td1 >> 16) & 0xFFFFu) | (32u << 16));         // tile_dim0 = 32
    g1[4] = (int)(unsigned)rows;                                  // tile_dim1, tile_dim2=0
    g1[5] = (int)(unsigned)K;                                     // tensor_dim0_stride lo
    g1[6] = 0;
    g1[7] = 0;
    i32x4 z4 = {0, 0, 0, 0};
#if __has_include(<hip/amd_detail/amd_gfx1250_TDM.h>)
    i32x8 z8 = {0, 0, 0, 0, 0, 0, 0, 0};
    __builtin_amdgcn_tensor_load_to_lds(g0, g1, z4, z4, z8, 0);
#else
    __builtin_amdgcn_tensor_load_to_lds(g0, g1, z4, z4, 0);
#endif
}
#endif // USE_TDM

// ---------------------------------------------------------------------------
// GEMM: out[M,N] = A[M,K](bf16) @ Wt[N,K](bf16, pre-transposed) + bias(f32)
// 128 threads = 4 waves in a 2x2 grid; block tile (2*WM)x(2*WN); each wave a
// WMxWN tile as (WM/16)x(WN/16) WMMA(16x16x32 bf16). K%32==0.
// Double-buffered LDS; TDM prefetches tile k+1 while tile k is computed.
// ---------------------------------------------------------------------------
template<int WM, int WN>
__global__ __launch_bounds__(128)
void gemm_bf16_kernel(const __bf16* __restrict__ A, const __bf16* __restrict__ Wt,
                      const float* __restrict__ bias, float* __restrict__ out,
                      int M, int N, int K)
{
    constexpr int BM = 2 * WM, BN = 2 * WN;
    constexpr int MI = WM / 16, NJ = WN / 16;
    __shared__ __attribute__((aligned(16))) __bf16 lA[2][BM * LDS_STRIDE];
    __shared__ __attribute__((aligned(16))) __bf16 lB[2][BN * LDS_STRIDE];

    const int tid  = threadIdx.x;
    const int lane = tid & 31;
    const int wave = tid >> 5;
    const int m0   = blockIdx.x * BM;
    const int n0   = blockIdx.y * BN;
    const int wm   = (wave & 1) * WM;
    const int wn   = (wave >> 1) * WN;
    const int lrow = lane & 15;
    // 16-bit WMMA operand layout: lanes 0-15 hold K{0..7,16..23},
    // lanes 16-31 hold K{8..15,24..31}
    const int kshift = (lane & 16) ? 8 : 0;

    v8f acc[MI][NJ];
#pragma unroll
    for (int i = 0; i < MI; ++i)
#pragma unroll
        for (int j = 0; j < NJ; ++j)
#pragma unroll
            for (int e = 0; e < 8; ++e) acc[i][j][e] = 0.0f;

    const int nIter = K >> 5;   // K/32 tiles

#if defined(USE_TDM)
    // prologue: tile 0 into buffer 0
    if (wave == 0) {
        tdm_load_tile(A  + (size_t)m0 * K, lA[0], BM, K, M);
        tdm_load_tile(Wt + (size_t)n0 * K, lB[0], BN, K, N);
    }
#endif

    for (int it = 0; it < nIter; ++it) {
        const int k0  = it << 5;
        const int cur = it & 1;
#if defined(USE_TDM)
        if (wave == 0) {
            if (it + 1 < nIter) {
                const int nxt = cur ^ 1;
                // issue DMA for tile it+1, then wait only for tile it's pair:
                // tensor ops complete in order, so <=2 outstanding means the
                // previous (A,B) pair has landed while the new pair flies.
                tdm_load_tile(A  + (size_t)m0 * K + k0 + 32, lA[nxt], BM, K, M);
                tdm_load_tile(Wt + (size_t)n0 * K + k0 + 32, lB[nxt], BN, K, N);
                __builtin_amdgcn_s_wait_tensorcnt(2);
            } else {
                __builtin_amdgcn_s_wait_tensorcnt(0);
            }
        }
        // prefetch two tiles ahead into caches (global_prefetch_b8)
        if (k0 + 64 < K) {
            if (tid < BM) __builtin_prefetch(A  + (size_t)(m0 + tid) * K + k0 + 64, 0, 3);
            if (tid < BN) __builtin_prefetch(Wt + (size_t)(n0 + tid) * K + k0 + 64, 0, 3);
        }
        __syncthreads();   // tile `it` visible to all waves
#else
        for (int c = tid; c < BM * 4; c += 128) {
            int r = c >> 2, c8 = c & 3;
            *(v8bf*)(lA[cur] + r * LDS_STRIDE + c8 * 8) =
                *(const v8bf*)(A + (size_t)(m0 + r) * K + k0 + c8 * 8);
        }
        for (int c = tid; c < BN * 4; c += 128) {
            int r = c >> 2, c8 = c & 3;
            *(v8bf*)(lB[cur] + r * LDS_STRIDE + c8 * 8) =
                *(const v8bf*)(Wt + (size_t)(n0 + r) * K + k0 + c8 * 8);
        }
        __syncthreads();
#endif
        v16bf aF[MI], bF[NJ];
#pragma unroll
        for (int i = 0; i < MI; ++i) {
            const v8bf* pa = (const v8bf*)(lA[cur] + (wm + i * 16 + lrow) * LDS_STRIDE + kshift);
            aF[i] = __builtin_shufflevector(pa[0], pa[2],
                    0,1,2,3,4,5,6,7,8,9,10,11,12,13,14,15);
        }
#pragma unroll
        for (int j = 0; j < NJ; ++j) {
            const v8bf* pb = (const v8bf*)(lB[cur] + (wn + j * 16 + lrow) * LDS_STRIDE + kshift);
            bF[j] = __builtin_shufflevector(pb[0], pb[2],
                    0,1,2,3,4,5,6,7,8,9,10,11,12,13,14,15);
        }
#pragma unroll
        for (int i = 0; i < MI; ++i)
#pragma unroll
            for (int j = 0; j < NJ; ++j)
                acc[i][j] = __builtin_amdgcn_wmma_f32_16x16x32_bf16(
                    false, aF[i], false, bF[j], (short)0, acc[i][j], false, false);
        // protect the buffer the next iteration's DMA/stage will overwrite
        __syncthreads();
    }

    // C layout: element r -> M = r (lanes 0-15) / 8+r (lanes 16-31); N = lane&15
#pragma unroll
    for (int i = 0; i < MI; ++i) {
#pragma unroll
        for (int j = 0; j < NJ; ++j) {
            int col   = n0 + wn + j * 16 + (lane & 15);
            int rbase = m0 + wm + i * 16 + ((lane >> 4) << 3);
            float bc  = bias[col];
#pragma unroll
            for (int r = 0; r < 8; ++r)
                out[(size_t)(rbase + r) * N + col] = acc[i][j][r] + bc;
        }
    }
}

// ---------------------------------------------------------------------------
// fp32 -> bf16 elementwise convert
__global__ __launch_bounds__(256)
void conv_bf16_kernel(const float* __restrict__ in, __bf16* __restrict__ out, int n)
{
    int i = blockIdx.x * blockDim.x + threadIdx.x;
    if (i < n) out[i] = f2bf(in[i]);
}

// fp32 W (K,N) -> bf16 Wt (N,K) transpose-convert
__global__ __launch_bounds__(256)
void trans_bf16_kernel(const float* __restrict__ W, __bf16* __restrict__ Wt, int K, int N)
{
    int i = blockIdx.x * blockDim.x + threadIdx.x;
    if (i < K * N) {
        int k = i / N, n = i - k * N;
        Wt[(size_t)n * K + k] = f2bf(W[i]);
    }
}

// LayerNorm over D=768 (fp32 in, bf16 out): one wave32 per row
__global__ __launch_bounds__(256)
void ln_bf16_kernel(const float* __restrict__ x, const float* __restrict__ g,
                    const float* __restrict__ beta, __bf16* __restrict__ y, int rows)
{
    const int wave = threadIdx.x >> 5;
    const int lane = threadIdx.x & 31;
    const int row  = blockIdx.x * 8 + wave;
    if (row >= rows) return;
    const float* xr = x + (size_t)row * 768;

    float s = 0.f, ss = 0.f;
    for (int c = lane; c < 768; c += 32) {
        float v = xr[c];
        s += v; ss += v * v;
    }
#pragma unroll
    for (int o = 16; o > 0; o >>= 1) {
        s  += __shfl_xor(s,  o, 32);
        ss += __shfl_xor(ss, o, 32);
    }
    float mean = s * (1.0f / 768.0f);
    float var  = ss * (1.0f / 768.0f) - mean * mean;
    float rstd = rsqrtf(var + 1e-6f);
    __bf16* yr = y + (size_t)row * 768;
    for (int c = lane; c < 768; c += 32) {
        float v = xr[c];
        yr[c] = f2bf((v - mean) * rstd * g[c] + beta[c]);
    }
}

// ---------------------------------------------------------------------------
// MS-Deformable-Attn sampling: one thread per (b, q, h, d) output element.
// value fp32 (B,Lf,768); off (B*Lq,128); aw (B*Lq,64) -> out bf16 (B,Lq,768)
__global__ __launch_bounds__(256)
void msda_kernel(const float* __restrict__ value, const float* __restrict__ off,
                 const float* __restrict__ aw, __bf16* __restrict__ out,
                 int Bn, int Lq, int Lf, int H, int W, int qe)
{
    int tid = blockIdx.x * blockDim.x + threadIdx.x;
    int total = Bn * Lq * 768;
    if (tid >= total) return;
    int d  = tid % 48;
    int h  = (tid / 48) & 15;
    int q  = (tid / 768) % Lq;
    int b  = tid / (768 * Lq);
    int bq = b * Lq + q;

    float rx = ((float)(q % qe) + 0.5f) / (float)qe;
    float ry = ((float)(q / qe) + 0.5f) / (float)qe;

    const float* awp = aw + (size_t)bq * 64 + h * 4;
    float l0 = awp[0], l1 = awp[1], l2 = awp[2], l3 = awp[3];
    float mx = fmaxf(fmaxf(l0, l1), fmaxf(l2, l3));
    float e0 = __expf(l0 - mx), e1 = __expf(l1 - mx);
    float e2 = __expf(l2 - mx), e3 = __expf(l3 - mx);
    float inv = 1.0f / (e0 + e1 + e2 + e3);
    float wgt[4] = { e0 * inv, e1 * inv, e2 * inv, e3 * inv };

    const float* offp = off + (size_t)bq * 128 + h * 8;
    const float* vb   = value + (size_t)b * Lf * 768 + h * 48 + d;

    float acc = 0.0f;
#pragma unroll
    for (int p = 0; p < 4; ++p) {
        float lx = rx + offp[p * 2 + 0] / (float)W;
        float ly = ry + offp[p * 2 + 1] / (float)H;
        float xf = lx * (float)W - 0.5f;
        float yf = ly * (float)H - 0.5f;
        float x0f = floorf(xf), y0f = floorf(yf);
        float wx1 = xf - x0f, wy1 = yf - y0f;
        int x0 = (int)x0f, y0 = (int)y0f;
        float s = 0.0f;
#pragma unroll
        for (int cy = 0; cy < 2; ++cy) {
#pragma unroll
            for (int cx = 0; cx < 2; ++cx) {
                int xi = x0 + cx, yi = y0 + cy;
                float w = (cx ? wx1 : 1.0f - wx1) * (cy ? wy1 : 1.0f - wy1);
                bool valid = (xi >= 0) & (xi < W) & (yi >= 0) & (yi < H);
                int xc = min(max(xi, 0), W - 1);
                int yc = min(max(yi, 0), H - 1);
                float v = vb[(size_t)(yc * W + xc) * 768];
                s += v * (valid ? w : 0.0f);
            }
        }
        acc += wgt[p] * s;
    }
    out[tid] = f2bf(acc);
}

// out = (attn + attn2) * gamma1
__global__ __launch_bounds__(256)
void combine_kernel(const float* __restrict__ a, const float* __restrict__ b,
                    const float* __restrict__ g, float* __restrict__ out, int total)
{
    int i = blockIdx.x * blockDim.x + threadIdx.x;
    if (i < total) out[i] = (a[i] + b[i]) * g[i % 768];
}

// ---------------------------------------------------------------------------
extern "C" void kernel_launch(void* const* d_in, const int* in_sizes, int n_in,
                              void* d_out, int out_size, void* d_ws, size_t ws_size,
                              hipStream_t stream)
{
    (void)in_sizes; (void)n_in; (void)out_size; (void)ws_size;

    const float* query   = (const float*)d_in[0];
    const float* feat    = (const float*)d_in[1];
    const float* qn_W    = (const float*)d_in[4];
    const float* qn_b    = (const float*)d_in[5];
    const float* qn_g    = (const float*)d_in[6];
    const float* qn_beta = (const float*)d_in[7];
    const float* fn_W    = (const float*)d_in[8];
    const float* fn_b    = (const float*)d_in[9];
    const float* fn_g    = (const float*)d_in[10];
    const float* fn_beta = (const float*)d_in[11];
    const float* nm_g    = (const float*)d_in[12];
    const float* nm_beta = (const float*)d_in[13];
    const float* gamma1  = (const float*)d_in[14];
    const float* c_Wv    = (const float*)d_in[15];
    const float* c_bv    = (const float*)d_in[16];
    const float* c_Woff  = (const float*)d_in[17];
    const float* c_boff  = (const float*)d_in[18];
    const float* c_Wa    = (const float*)d_in[19];
    const float* c_ba    = (const float*)d_in[20];
    const float* c_Wo    = (const float*)d_in[21];
    const float* c_bo    = (const float*)d_in[22];
    const float* s_Wv    = (const float*)d_in[23];
    const float* s_bv    = (const float*)d_in[24];
    const float* s_Woff  = (const float*)d_in[25];
    const float* s_boff  = (const float*)d_in[26];
    const float* s_Wa    = (const float*)d_in[27];
    const float* s_ba    = (const float*)d_in[28];
    const float* s_Wo    = (const float*)d_in[29];
    const float* s_bo    = (const float*)d_in[30];

    const int Bn = 2, Lq = 4096, Lf = 4096, D = 768, K = 768;
    const int M = Bn * Lq;                 // 8192
    const int total = M * D;
    const size_t BIGF = (size_t)M * D;     // fp32 elements
    const size_t BIGH = (size_t)M * K;     // bf16 elements

    float*  fT    = (float*)d_ws;          // gemm fp32 scratch / attn2
    float*  fV    = fT + BIGF;             // value
    float*  fAttn = fV + BIGF;             // attn (kept for final combine)
    float*  offb  = fAttn + BIGF;          // (M,128)
    float*  awb   = offb + (size_t)M * 128;// (M,64)
    __bf16* hA    = (__bf16*)(awb + (size_t)M * 64);  // query-bf / qn / attn1
    __bf16* hB    = hA + BIGH;                        // feat-bf / fn / samp
    __bf16* wt    = hB + BIGH;                        // transposed weight (<=768x768)

    dim3 blk(128), blk256(256);
    dim3 gBig(M / 128, D / 128);   // N=768
    dim3 gOff(M / 128, 1);         // N=128 (one 128-wide column block)
    dim3 gAw(M / 128, 1);          // N=64  (WN=32 config)
    int lnBlocks = M / 8;
    int elBlocks = (total + 255) / 256;
    int cvBlocks = (int)((BIGH + 255) / 256);
    int twBlocks = (K * D + 255) / 256;    // 768x768 transpose
    int toBlocks = (K * 128 + 255) / 256;
    int taBlocks = (K * 64 + 255) / 256;

    // bf16 copies of the two fp32 inputs
    conv_bf16_kernel<<<cvBlocks, blk256, 0, stream>>>(query, hA, (int)BIGH);
    conv_bf16_kernel<<<cvBlocks, blk256, 0, stream>>>(feat,  hB, (int)BIGH);

    // qn = LN(query @ qn_W + qn_b) -> hA (bf16)
    trans_bf16_kernel<<<twBlocks, blk256, 0, stream>>>(qn_W, wt, K, D);
    gemm_bf16_kernel<64,64><<<gBig, blk, 0, stream>>>(hA, wt, qn_b, fT, M, D, K);
    ln_bf16_kernel<<<lnBlocks, blk256, 0, stream>>>(fT, qn_g, qn_beta, hA, M);
    // fn = LN(feat @ fn_W + fn_b) -> hB (bf16)
    trans_bf16_kernel<<<twBlocks, blk256, 0, stream>>>(fn_W, wt, K, D);
    gemm_bf16_kernel<64,64><<<gBig, blk, 0, stream>>>(hB, wt, fn_b, fT, M, D, K);
    ln_bf16_kernel<<<lnBlocks, blk256, 0, stream>>>(fT, fn_g, fn_beta, hB, M);

    // --- cross deformable attention ---
    trans_bf16_kernel<<<twBlocks, blk256, 0, stream>>>(c_Wv, wt, K, D);
    gemm_bf16_kernel<64,64><<<gBig, blk, 0, stream>>>(hB, wt, c_bv, fV, M, D, K);
    trans_bf16_kernel<<<toBlocks, blk256, 0, stream>>>(c_Woff, wt, K, 128);
    gemm_bf16_kernel<64,64><<<gOff, blk, 0, stream>>>(hA, wt, c_boff, offb, M, 128, K);
    trans_bf16_kernel<<<taBlocks, blk256, 0, stream>>>(c_Wa, wt, K, 64);
    gemm_bf16_kernel<64,32><<<gAw, blk, 0, stream>>>(hA, wt, c_ba, awb, M, 64, K);
    msda_kernel<<<elBlocks, blk256, 0, stream>>>(fV, offb, awb, hB, Bn, Lq, Lf, 64, 64, 64);
    trans_bf16_kernel<<<twBlocks, blk256, 0, stream>>>(c_Wo, wt, K, D);
    gemm_bf16_kernel<64,64><<<gBig, blk, 0, stream>>>(hB, wt, c_bo, fAttn, M, D, K);

    // attn1 = LN(attn) -> hA (bf16); fAttn preserved
    ln_bf16_kernel<<<lnBlocks, blk256, 0, stream>>>(fAttn, nm_g, nm_beta, hA, M);

    // --- self deformable attention ---
    trans_bf16_kernel<<<twBlocks, blk256, 0, stream>>>(s_Wv, wt, K, D);
    gemm_bf16_kernel<64,64><<<gBig, blk, 0, stream>>>(hA, wt, s_bv, fV, M, D, K);
    trans_bf16_kernel<<<toBlocks, blk256, 0, stream>>>(s_Woff, wt, K, 128);
    gemm_bf16_kernel<64,64><<<gOff, blk, 0, stream>>>(hA, wt, s_boff, offb, M, 128, K);
    trans_bf16_kernel<<<taBlocks, blk256, 0, stream>>>(s_Wa, wt, K, 64);
    gemm_bf16_kernel<64,32><<<gAw, blk, 0, stream>>>(hA, wt, s_ba, awb, M, 64, K);
    msda_kernel<<<elBlocks, blk256, 0, stream>>>(fV, offb, awb, hB, Bn, Lq, Lq, 64, 64, 64);
    trans_bf16_kernel<<<twBlocks, blk256, 0, stream>>>(s_Wo, wt, K, D);
    gemm_bf16_kernel<64,64><<<gBig, blk, 0, stream>>>(hB, wt, s_bo, fT, M, D, K);

    // out = (attn + attn2) * gamma1
    combine_kernel<<<elBlocks, blk256, 0, stream>>>(fAttn, fT, gamma1, (float*)d_out, total);
}
